// RefinedCodeSwitchLoss_6914897346751
// MI455X (gfx1250) — matrix operand order repaired and checked by reading, not compiled, not run on hardware
//
#include <hip/hip_runtime.h>
#include <cstdint>
#include <cstddef>

// ---- problem constants ----
#define B_ROWS 2048
#define DIM    1024
#define NQ     (2*B_ROWS)          // 4096 query rows ([eng;kor])
#define NK     (4*B_ROWS)          // 8192 key rows (all4)
#define NSPLIT 16                  // column splits of the fused GEMM
#define COLS_PER_SPLIT (NK/NSPLIT) // 512
#define MT     128                 // query rows per workgroup
#define NB     128                 // key cols per n-block
#define KSTEP  64                  // bf16 K per LDS stage (2 WMMA K-steps)
#define LDSTR  72                  // 64 + 8 pad (bf16) -> 144B rows: 16B aligned, conflict-free

typedef __attribute__((ext_vector_type(8)))  float  v8f;
typedef __attribute__((ext_vector_type(16))) __bf16 v16bf;

union Frag {
    v16bf v;
    uint4 q[2];
};

__device__ __forceinline__ unsigned short f2bf(float f) {
    // round-to-nearest-even float -> bf16 (inputs are finite)
    unsigned int u = __float_as_uint(f);
    u += 0x7FFFu + ((u >> 16) & 1u);
    return (unsigned short)(u >> 16);
}

// LDS byte offset of a __shared__ object: generic->LDS addrspacecast is a
// low-32-bit truncation on amdgcn.
__device__ __forceinline__ unsigned lds_off(const void* p) {
    return (unsigned)(uintptr_t)p;
}

// Direct global -> LDS async copy, 16 bytes per lane (ASYNCcnt-tracked).
__device__ __forceinline__ void async_copy_b128(unsigned lds_addr, const void* gptr) {
    asm volatile("global_load_async_to_lds_b128 %0, %1, off"
                 :: "v"(lds_addr), "v"((unsigned long long)(uintptr_t)gptr)
                 : "memory");
}

__device__ __forceinline__ void wait_async0() {
    asm volatile("s_wait_asynccnt 0x0" ::: "memory");
}

// Hard scheduling fence: nothing may be reordered across this point.
__device__ __forceinline__ void sched_fence() {
#if __has_builtin(__builtin_amdgcn_sched_barrier)
    __builtin_amdgcn_sched_barrier(0);
#else
    asm volatile("" ::: "memory");
#endif
}

// ---------------------------------------------------------------------------
// Kernel A: per-row L2 normalize all four matrices, emit bf16 all4 + inv norms
// all4 row order matches reference: [eng, kor, etok, ktoe]
// ---------------------------------------------------------------------------
__global__ __launch_bounds__(128) void norm_kernel(
    const float* __restrict__ eng,  const float* __restrict__ etok,
    const float* __restrict__ ktoe, const float* __restrict__ kor,
    unsigned short* __restrict__ all4, float* __restrict__ inv_norm)
{
    const int j   = blockIdx.x;          // 0..8191
    const int tid = threadIdx.x;         // 0..127
    const float* src;
    if      (j < 2048) src = eng  + (size_t)j        * DIM;
    else if (j < 4096) src = kor  + (size_t)(j-2048) * DIM;
    else if (j < 6144) src = etok + (size_t)(j-4096) * DIM;
    else               src = ktoe + (size_t)(j-6144) * DIM;

    const float4* s4 = (const float4*)src;
    float4 x0 = s4[2*tid + 0];
    float4 x1 = s4[2*tid + 1];
    float vals[8] = {x0.x, x0.y, x0.z, x0.w, x1.x, x1.y, x1.z, x1.w};

    float ss = 0.f;
    #pragma unroll
    for (int e = 0; e < 8; ++e) ss += vals[e] * vals[e];
    #pragma unroll
    for (int m = 1; m < 32; m <<= 1) ss += __shfl_xor(ss, m, 32);

    __shared__ float wsum[4];
    if ((tid & 31) == 0) wsum[tid >> 5] = ss;
    __syncthreads();
    const float tot = wsum[0] + wsum[1] + wsum[2] + wsum[3];
    const float inv = 1.0f / sqrtf(tot);
    if (tid == 0) inv_norm[j] = inv;

    unsigned int w0 = (unsigned)f2bf(vals[0]*inv) | ((unsigned)f2bf(vals[1]*inv) << 16);
    unsigned int w1 = (unsigned)f2bf(vals[2]*inv) | ((unsigned)f2bf(vals[3]*inv) << 16);
    unsigned int w2 = (unsigned)f2bf(vals[4]*inv) | ((unsigned)f2bf(vals[5]*inv) << 16);
    unsigned int w3 = (unsigned)f2bf(vals[6]*inv) | ((unsigned)f2bf(vals[7]*inv) << 16);
    ((uint4*)(all4 + (size_t)j * DIM))[tid] = make_uint4(w0, w1, w2, w3);
}

// ---------------------------------------------------------------------------
// Kernel B: per-row i (0..2047): positive-pair dots (f32), lognum via LSE3,
// cs_reg row norms.
// ---------------------------------------------------------------------------
__global__ __launch_bounds__(128) void rowstats_kernel(
    const float* __restrict__ eng,  const float* __restrict__ etok,
    const float* __restrict__ ktoe, const float* __restrict__ kor,
    const float* __restrict__ ratio_etok, const float* __restrict__ ratio_ktoe,
    const float* __restrict__ inv_norm,
    float* __restrict__ lognum_e, float* __restrict__ lognum_k,
    float* __restrict__ csreg)
{
    const int i   = blockIdx.x;
    const int tid = threadIdx.x;

    const float inv_e  = inv_norm[i];
    const float inv_k  = inv_norm[2048 + i];
    const float inv_tk = inv_norm[4096 + i];
    const float inv_te = inv_norm[6144 + i];
    const float r1 = ratio_etok[i];
    const float r2 = ratio_ktoe[i];

    const float4* pe = (const float4*)(eng  + (size_t)i * DIM);
    const float4* pk = (const float4*)(kor  + (size_t)i * DIM);
    const float4* pt = (const float4*)(etok + (size_t)i * DIM);
    const float4* pu = (const float4*)(ktoe + (size_t)i * DIM);

    float4 e0 = pe[2*tid], e1 = pe[2*tid+1];
    float4 k0 = pk[2*tid], k1 = pk[2*tid+1];
    float4 t0 = pt[2*tid], t1 = pt[2*tid+1];
    float4 u0 = pu[2*tid], u1 = pu[2*tid+1];
    float ev[8] = {e0.x,e0.y,e0.z,e0.w,e1.x,e1.y,e1.z,e1.w};
    float kv[8] = {k0.x,k0.y,k0.z,k0.w,k1.x,k1.y,k1.z,k1.w};
    float tv[8] = {t0.x,t0.y,t0.z,t0.w,t1.x,t1.y,t1.z,t1.w};
    float uv[8] = {u0.x,u0.y,u0.z,u0.w,u1.x,u1.y,u1.z,u1.w};

    // acc: 0:e.tk 1:e.te 2:e.k 3:k.tk 4:k.te 5:s1 6:s2
    float acc[7] = {0,0,0,0,0,0,0};
    #pragma unroll
    for (int e = 0; e < 8; ++e) {
        const float en  = ev[e] * inv_e;
        const float kn  = kv[e] * inv_k;
        const float tkn = tv[e] * inv_tk;
        const float ten = uv[e] * inv_te;
        acc[0] += en * tkn;
        acc[1] += en * ten;
        acc[2] += en * kn;
        acc[3] += kn * tkn;
        acc[4] += kn * ten;
        const float z1 = tkn - (r1 * en + (1.0f - r1) * kn);
        acc[5] += z1 * z1;
        const float z2 = ten - (r2 * kn + (1.0f - r2) * en);
        acc[6] += z2 * z2;
    }
    #pragma unroll
    for (int c = 0; c < 7; ++c) {
        float x = acc[c];
        #pragma unroll
        for (int m = 1; m < 32; m <<= 1) x += __shfl_xor(x, m, 32);
        acc[c] = x;
    }
    __shared__ float red[4][7];
    if ((tid & 31) == 0) {
        #pragma unroll
        for (int c = 0; c < 7; ++c) red[tid >> 5][c] = acc[c];
    }
    __syncthreads();
    if (tid == 0) {
        float t[7];
        #pragma unroll
        for (int c = 0; c < 7; ++c) t[c] = red[0][c] + red[1][c] + red[2][c] + red[3][c];
        // pos_eng = 10*{e.tk, e.te, e.k}; pos_kor = 10*{k.tk, k.te, e.k}
        float p0 = 10.f * t[0], p1 = 10.f * t[1], p2 = 10.f * t[2];
        float me = fmaxf(p0, fmaxf(p1, p2));
        lognum_e[i] = me + __logf(__expf(p0-me) + __expf(p1-me) + __expf(p2-me));
        float q0 = 10.f * t[3], q1 = 10.f * t[4], q2 = 10.f * t[2];
        float mk = fmaxf(q0, fmaxf(q1, q2));
        lognum_k[i] = mk + __logf(__expf(q0-mk) + __expf(q1-mk) + __expf(q2-mk));
        csreg[i] = sqrtf(t[5]) + sqrtf(t[6]);
    }
}

// ---------------------------------------------------------------------------
// Kernel C: fused WMMA GEMM + exp + self-mask + per-row partial sums.
// Grid: (NSPLIT, NQ/MT). Block: 256 threads = 8 wave32; wave w owns rows
// [rbase+16w, +16). Double-buffered LDS tiles filled with
// global_load_async_to_lds_b128; one workgroup barrier per K-stage.
// A sched_barrier(0) pins all 18 ds_load_b128 of a K-substep ahead of the
// 8 WMMAs, so waits retire with loads still in flight (no dscnt-0 stalls).
// partial[row][split] = sum_{cols in split, col != row} exp(10*dot - 10)
// ---------------------------------------------------------------------------
__global__ __launch_bounds__(256) void simexp_kernel(
    const unsigned short* __restrict__ all4, float* __restrict__ partial)
{
    // [buffer][A=0/B=1][row*LDSTR + col]
    __shared__ __align__(16) unsigned short Sh[2][2][MT * LDSTR];

    const int s     = blockIdx.x;           // column split 0..NSPLIT-1
    const int mt    = blockIdx.y;           // row tile 0..31
    const int rbase = mt * MT;
    const int tid   = threadIdx.x;
    const int w     = tid >> 5;             // wave 0..7
    const int lane  = tid & 31;
    const int half  = lane >> 4;            // 0 or 1
    const int mrow  = lane & 15;            // N position in C fragment

    const int srow = tid >> 1;              // staging row 0..127
    const int scol = (tid & 1) * 32;        // staging col 0 or 32 (elems)

    float rowAcc[8];
    #pragma unroll
    for (int v = 0; v < 8; ++v) rowAcc[v] = 0.f;

    for (int nb = 0; nb < COLS_PER_SPLIT / NB; ++nb) {
        const int cbase = s * COLS_PER_SPLIT + nb * NB;
        v8f acc[8];
        v8f zero = {};
        #pragma unroll
        for (int t = 0; t < 8; ++t) acc[t] = zero;

        // make sure previous stage's LDS consumers are done before refilling
        __syncthreads();

        // prologue: stage K-tile 0 into buffer 0
        {
            const unsigned short* ga = all4 + (size_t)(rbase + srow) * DIM + scol;
            const unsigned short* gb = all4 + (size_t)(cbase + srow) * DIM + scol;
            const unsigned la = lds_off(&Sh[0][0][srow * LDSTR + scol]);
            const unsigned lb = lds_off(&Sh[0][1][srow * LDSTR + scol]);
            #pragma unroll
            for (int c = 0; c < 4; ++c) {
                async_copy_b128(la + c * 16, ga + c * 8);
                async_copy_b128(lb + c * 16, gb + c * 8);
            }
        }

        int cur = 0;
        for (int kk = 0; kk < DIM; kk += KSTEP) {
            wait_async0();          // my buf[cur] copies have landed
            __syncthreads();        // everyone's buf[cur] copies have landed

            if (kk + KSTEP < DIM) { // prefetch next K-tile into the other buffer
                const int nxt = cur ^ 1;
                const unsigned short* ga = all4 + (size_t)(rbase + srow) * DIM + (kk + KSTEP) + scol;
                const unsigned short* gb = all4 + (size_t)(cbase + srow) * DIM + (kk + KSTEP) + scol;
                const unsigned la = lds_off(&Sh[nxt][0][srow * LDSTR + scol]);
                const unsigned lb = lds_off(&Sh[nxt][1][srow * LDSTR + scol]);
                #pragma unroll
                for (int c = 0; c < 4; ++c) {
                    async_copy_b128(la + c * 16, ga + c * 8);
                    async_copy_b128(lb + c * 16, gb + c * 8);
                }
            }

            const unsigned short* At = &Sh[cur][0][0];
            const unsigned short* Bt = &Sh[cur][1][0];
            #pragma unroll
            for (int ks = 0; ks < KSTEP; ks += 32) {
                // A fragment: row = lane%16 of wave's 16-row band;
                // K runs [ks+8*half, +8) and [ks+16+8*half, +8)
                Frag a;
                const int abase = (16 * w + mrow) * LDSTR + ks + half * 8;
                a.q[0] = *(const uint4*)&At[abase];
                a.q[1] = *(const uint4*)&At[abase + 16];

                // batch-load all 8 B fragments (16 ds_load_b128 in flight)
                Frag b[8];
                #pragma unroll
                for (int t = 0; t < 8; ++t) {
                    const int bbase = (16 * t + mrow) * LDSTR + ks + half * 8;
                    b[t].q[0] = *(const uint4*)&Bt[bbase];
                    b[t].q[1] = *(const uint4*)&Bt[bbase + 16];
                }
                // pin the loads ahead of the WMMAs: forces 8 live fragments
                // and descending s_wait_dscnt instead of 0-waits
                sched_fence();
                #pragma unroll
                for (int t = 0; t < 8; ++t) {
                    acc[t] = __builtin_amdgcn_wmma_f32_16x16x32_bf16(
                        false, a.v, false, b[t].v, (short)0, acc[t], false, false);
                }
            }
            cur ^= 1;
        }

        // post-process: sim = 10*dot; accumulate exp(sim-10), masking col==row
        #pragma unroll
        for (int t = 0; t < 8; ++t) {
            const int gcol = cbase + 16 * t + mrow;
            #pragma unroll
            for (int v = 0; v < 8; ++v) {
                const int grow = rbase + 16 * w + v + 8 * half;  // C layout: M = v + 8*half
                float e = __expf(acc[t][v] * 10.0f - 10.0f);
                if (gcol == grow) e = 0.0f;                       // exclude self column
                rowAcc[v] += e;
            }
        }
    }

    // reduce across the 16 lanes (N positions) of each half-wave
    #pragma unroll
    for (int v = 0; v < 8; ++v) {
        float x = rowAcc[v];
        x += __shfl_xor(x, 1, 32);
        x += __shfl_xor(x, 2, 32);
        x += __shfl_xor(x, 4, 32);
        x += __shfl_xor(x, 8, 32);
        rowAcc[v] = x;
    }
    if (mrow == 0) {  // lanes 0 (rows M=0..7) and 16 (rows M=8..15)
        #pragma unroll
        for (int v = 0; v < 8; ++v) {
            const int grow = rbase + 16 * w + v + 8 * half;
            partial[(size_t)grow * NSPLIT + s] = rowAcc[v];
        }
    }
}

// ---------------------------------------------------------------------------
// Kernel D: deterministic final reduction -> scalar loss
// ---------------------------------------------------------------------------
__global__ __launch_bounds__(256) void finalize_kernel(
    const float* __restrict__ partial,
    const float* __restrict__ lognum_e, const float* __restrict__ lognum_k,
    const float* __restrict__ csreg,
    float* __restrict__ logden, float* __restrict__ out)
{
    const int tid = threadIdx.x;
    for (int r = tid; r < NQ; r += 256) {
        float ssum = 0.f;
        #pragma unroll
        for (int j = 0; j < NSPLIT; ++j) ssum += partial[(size_t)r * NSPLIT + j];
        logden[r] = 10.0f + __logf(ssum);
    }
    __syncthreads();

    float contr = 0.f, cs = 0.f;
    for (int i = tid; i < B_ROWS; i += 256) {
        contr += (logden[i]          - lognum_e[i])
               + (logden[B_ROWS + i] - lognum_k[i]);
        cs += csreg[i];
    }
    __shared__ float rc[256], rs[256];
    rc[tid] = contr;
    rs[tid] = cs;
    __syncthreads();
    for (int st = 128; st > 0; st >>= 1) {
        if (tid < st) { rc[tid] += rc[tid + st]; rs[tid] += rs[tid + st]; }
        __syncthreads();
    }
    if (tid == 0)
        out[0] = rc[0] / (float)NQ + 0.5f * rs[0] / (float)B_ROWS;
}

// ---------------------------------------------------------------------------
extern "C" void kernel_launch(void* const* d_in, const int* in_sizes, int n_in,
                              void* d_out, int out_size, void* d_ws, size_t ws_size,
                              hipStream_t stream)
{
    (void)in_sizes; (void)n_in; (void)out_size; (void)ws_size;
    const float* eng   = (const float*)d_in[0];
    const float* etok  = (const float*)d_in[1];
    const float* ktoe  = (const float*)d_in[2];
    const float* kor   = (const float*)d_in[3];
    const float* r_etk = (const float*)d_in[4];
    const float* r_kte = (const float*)d_in[5];

    char* ws = (char*)d_ws;
    size_t off = 0;
    unsigned short* all4 = (unsigned short*)(ws + off); off += (size_t)NK * DIM * sizeof(unsigned short); // 16 MB
    float* inv_norm = (float*)(ws + off); off += (size_t)NK * sizeof(float);
    float* partial  = (float*)(ws + off); off += (size_t)NQ * NSPLIT * sizeof(float);
    float* lognum_e = (float*)(ws + off); off += (size_t)B_ROWS * sizeof(float);
    float* lognum_k = (float*)(ws + off); off += (size_t)B_ROWS * sizeof(float);
    float* csreg    = (float*)(ws + off); off += (size_t)B_ROWS * sizeof(float);
    float* logden   = (float*)(ws + off); off += (size_t)NQ * sizeof(float);

    norm_kernel<<<NK, 128, 0, stream>>>(eng, etok, ktoe, kor, all4, inv_norm);
    rowstats_kernel<<<B_ROWS, 128, 0, stream>>>(eng, etok, ktoe, kor,
                                                r_etk, r_kte, inv_norm,
                                                lognum_e, lognum_k, csreg);
    dim3 grid(NSPLIT, NQ / MT);   // 16 x 32 = 512 workgroups
    simexp_kernel<<<grid, 256, 0, stream>>>(all4, partial);
    finalize_kernel<<<1, 256, 0, stream>>>(partial, lognum_e, lognum_k, csreg,
                                           logden, (float*)d_out);
}
